// CNN_78975858639488
// MI455X (gfx1250) — compile-verified, matrix operand order checked by reading, not run on hardware
//
#include <hip/hip_runtime.h>
#include <math.h>

#define EPS 1e-5f

// ---------- CDNA5 WMMA types ----------
typedef __attribute__((ext_vector_type(16))) __bf16 v16bf;
typedef __attribute__((ext_vector_type(8)))  float  v8f;

union Frag16 { v16bf v; unsigned short u[16]; };

// ---------- address-space helper types ----------
typedef __attribute__((address_space(1))) int GInt;
typedef __attribute__((address_space(3))) int LInt;

// ---------- optional CDNA5 async global->LDS path (probe via __has_builtin) ----------
#if __has_builtin(__builtin_amdgcn_global_load_async_to_lds_b32) && \
    __has_builtin(__builtin_amdgcn_s_wait_asynccnt)
#define ATHENA_ASYNC_LDS 1
#else
#define ATHENA_ASYNC_LDS 0
#endif

// ---------- optional CDNA5 Tensor Data Mover path ----------
#if __has_builtin(__builtin_amdgcn_tensor_load_to_lds) && \
    __has_builtin(__builtin_amdgcn_s_wait_tensorcnt)
#define ATHENA_TDM 1
typedef __attribute__((ext_vector_type(4))) unsigned int u32x4;
typedef __attribute__((ext_vector_type(8))) int          i32x8;
typedef __attribute__((ext_vector_type(4))) int          i32x4;
#else
#define ATHENA_TDM 0
#endif

// ---------- bf16 bit helpers (round-to-nearest-even) ----------
__device__ __forceinline__ unsigned short f2bf(float f) {
    unsigned u = __float_as_uint(f);
    unsigned r = u + 0x7fffu + ((u >> 16) & 1u);
    return (unsigned short)(r >> 16);
}
__device__ __forceinline__ float bf2f(unsigned short h) {
    return __uint_as_float(((unsigned)h) << 16);
}

// ---------- small utility kernels ----------
__global__ void zero_f(float* p, int n) {
    int i = blockIdx.x * blockDim.x + threadIdx.x;
    if (i < n) p[i] = 0.f;
}

__global__ __launch_bounds__(256) void reduce_sum_sq(const float* __restrict__ x,
                                                     float* stats, int N) {
    __shared__ float s1[256], s2[256];
    float a = 0.f, b = 0.f;
    for (int i = blockIdx.x * blockDim.x + threadIdx.x; i < N; i += gridDim.x * blockDim.x) {
        float v = x[i]; a += v; b += v * v;
    }
    s1[threadIdx.x] = a; s2[threadIdx.x] = b;
    __syncthreads();
    for (int s = 128; s > 0; s >>= 1) {
        if (threadIdx.x < s) { s1[threadIdx.x] += s1[threadIdx.x + s]; s2[threadIdx.x] += s2[threadIdx.x + s]; }
        __syncthreads();
    }
    if (threadIdx.x == 0) { atomicAdd(&stats[0], s1[0]); atomicAdd(&stats[1], s2[0]); }
}

__global__ __launch_bounds__(256) void normalize_x(const float* __restrict__ x,
                                                   const float* __restrict__ stats,
                                                   unsigned short* __restrict__ out, int N) {
    float m = stats[0] / (float)N;
    float var = stats[1] / (float)N - m * m;
    float inv = rsqrtf(var + EPS);
    for (int i = blockIdx.x * blockDim.x + threadIdx.x; i < N; i += gridDim.x * blockDim.x)
        out[i] = f2bf((x[i] - m) * inv);
}

// ---------- MT19937 + numpy legacy polar gauss (basis replication) ----------
struct MTState { unsigned mt[624]; int mti; int hasg; double g; };

__device__ void mt_seed(MTState& s, unsigned seed) {
    s.mt[0] = seed;
    for (int i = 1; i < 624; i++)
        s.mt[i] = 1812433253u * (s.mt[i - 1] ^ (s.mt[i - 1] >> 30)) + (unsigned)i;
    s.mti = 624; s.hasg = 0; s.g = 0.0;
}
__device__ unsigned mt_u32(MTState& s) {
    if (s.mti >= 624) {
        for (int i = 0; i < 624; i++) {
            unsigned y = (s.mt[i] & 0x80000000u) | (s.mt[(i + 1) % 624] & 0x7fffffffu);
            s.mt[i] = s.mt[(i + 397) % 624] ^ (y >> 1) ^ ((y & 1u) ? 2567483615u : 0u);
        }
        s.mti = 0;
    }
    unsigned y = s.mt[s.mti++];
    y ^= y >> 11; y ^= (y << 7) & 2636928640u; y ^= (y << 15) & 4022730752u; y ^= y >> 18;
    return y;
}
__device__ double mt_dbl(MTState& s) {
    unsigned a = mt_u32(s) >> 5, b = mt_u32(s) >> 6;
    return ((double)a * 67108864.0 + (double)b) / 9007199254740992.0;
}
__device__ double mt_gauss(MTState& s) {
    if (s.hasg) { s.hasg = 0; return s.g; }
    double f, x1, x2, r2;
    do {
        x1 = 2.0 * mt_dbl(s) - 1.0;
        x2 = 2.0 * mt_dbl(s) - 1.0;
        r2 = x1 * x1 + x2 * x2;
    } while (r2 >= 1.0 || r2 == 0.0);
    f = sqrt(-2.0 * log(r2) / r2);
    s.g = f * x1; s.hasg = 1;
    return f * x2;
}

__global__ void gen_basis(unsigned seed, int total, float scale, float* __restrict__ out) {
    if (threadIdx.x == 0 && blockIdx.x == 0) {
        MTState s;
        mt_seed(s, seed);
        for (int i = 0; i < total; i++)
            out[i] = (float)mt_gauss(s) * scale;
    }
}

// ---------- kernel assembly: einsum('oin,nabxyz->oaibxyz') block into kern ----------
__global__ __launch_bounds__(256) void assemble(const float* __restrict__ w,
                                                const float* __restrict__ basis,
                                                float* __restrict__ kern,
                                                int mo, int dO, int mi, int dI, int nb,
                                                int row0, int col0, int Cin) {
    int total = mo * dO * mi * dI * 343;
    for (int idx = blockIdx.x * blockDim.x + threadIdx.x; idx < total; idx += gridDim.x * blockDim.x) {
        int r = idx % 343; int t = idx / 343;
        int b = t % dI; t /= dI;
        int i = t % mi; t /= mi;
        int a = t % dO; t /= dO;
        int o = t;
        float acc = 0.f;
        for (int n = 0; n < nb; n++)
            acc += w[(o * mi + i) * nb + n] * basis[((n * dO + a) * dI + b) * 343 + r];
        kern[((size_t)(row0 + o * dO + a) * Cin + (col0 + i * dI + b)) * 343 + r] = acc;
    }
}

// ---------- fp32 kernel -> bf16 GEMM-A, padded pow2 K layout ----------
// k = ci*512 + kx*64 + ky*8 + kz ; positions with kx/ky/kz==7 (pad) hold zero.
__global__ __launch_bounds__(256) void kern_to_bf16(const float* __restrict__ kern,
                                                    unsigned short* __restrict__ A,
                                                    int Cout, int Cin, int Kpad, int CoutPad) {
    int total = CoutPad * Kpad;
    for (int idx = blockIdx.x * blockDim.x + threadIdx.x; idx < total; idx += gridDim.x * blockDim.x) {
        int c = idx / Kpad, k = idx - c * Kpad;
        int ci = k >> 9;
        int kx = (k >> 6) & 7, ky = (k >> 3) & 7, kz = k & 7;
        float v = 0.f;
        if (c < Cout && kx < 7 && ky < 7 && kz < 7)
            v = kern[((size_t)c * Cin + ci) * 343 + kx * 49 + ky * 7 + kz];
        A[idx] = f2bf(v);
    }
}

// ---------- tensor-feature expansion (v otimes v), NCDHW bf16 ----------
__global__ __launch_bounds__(256) void build_ext(const unsigned short* __restrict__ act,
                                                 unsigned short* __restrict__ ext,
                                                 int B, int C, int si, int vi, int DHW) {
    int Cext = C + 9 * vi;
    int total = B * Cext * DHW;
    for (int idx = blockIdx.x * blockDim.x + threadIdx.x; idx < total; idx += gridDim.x * blockDim.x) {
        int vox = idx % DHW; int t = idx / DHW;
        int c = t % Cext; int n = t / Cext;
        unsigned short o;
        if (c < C) {
            o = act[(size_t)(n * C + c) * DHW + vox];
        } else {
            int tt = c - C; int v = tt / 9; int ij = tt - v * 9;
            int i = ij / 3, j = ij - (ij / 3) * 3;
            float a = bf2f(act[(size_t)(n * C + si + 3 * v + i) * DHW + vox]);
            float b = bf2f(act[(size_t)(n * C + si + 3 * v + j) * DHW + vox]);
            o = f2bf(a * b);
        }
        ext[idx] = o;
    }
}

// ---------- implicit-GEMM conv via v_wmma_f32_16x16x32_bf16 ----------
// A = weights [CoutPad][Kpad] bf16 (pow2 K layout), B gathered from ext (NCDHW bf16).
// 256 thr = 8 waves; each wave computes MT 16x16 cout tiles for its 16 voxels,
// reusing one gathered B fragment per k-step.
// A tile staging: MT==4 uses the Tensor Data Mover (one 2D-tile DMA per k-step,
// issued by wave 0, published via s_wait_tensorcnt + barrier); otherwise
// cooperative async global->LDS b32 transfers.
template <int MT>
__global__ __launch_bounds__(256) void conv_wmma(
    const unsigned short* __restrict__ ext, const unsigned short* __restrict__ A,
    float* __restrict__ y,
    int Cin, int ID, int IH, int IW,
    int OD, int OH, int OW, int Kpad, int Nvox)
{
    __shared__ unsigned short sA[MT * 16 * 32];
    const int tid  = threadIdx.x;
    const int lane = tid & 31;
    const int wave = tid >> 5;
    const int voxBase = blockIdx.x * 128 + wave * 16;

    // this lane's B column (voxel) coordinates -- Nvox is a multiple of 128
    const int col = lane & 15;
    const int q = voxBase + col;
    const int ODHW = OD * OH * OW;
    int n  = q / ODHW;       int r0 = q - n * ODHW;
    int od = r0 / (OH * OW); int r1 = r0 - od * (OH * OW);
    int oh = r1 / OW;        int ow = r1 - oh * OW;
    const int idb = od * 2 - 3, ihb = oh * 2 - 3, iwb = ow * 2 - 3;
    const int IHIW = IH * IW;
    const int IDHW = ID * IHIW;
    const int base = n * Cin * IDHW + idb * IHIW + ihb * IW + iwb;

    // K sub-index map for 16-bit A/B fragments (ISA 7.12.2 layout)
    int kk[8];
#pragma unroll
    for (int v = 0; v < 8; ++v)
        kk[v] = ((v < 4) ? (2 * v) : (16 + 2 * (v - 4))) + ((lane < 16) ? 0 : 8);
    const int row = lane & 15;

    v8f c[MT] = {};
    const int ar = tid >> 4;          // 0..15
    const int ak = (tid & 15) * 2;    // 0..30 even

#if ATHENA_TDM
    const unsigned ldsAddr = (unsigned)(size_t)(LInt*)&sA[0];
    const unsigned long long gbase = (unsigned long long)(size_t)A;
#endif

    for (int ks = 0; ks < Kpad; ks += 32) {
        // ---- stage A tiles (MT*16 rows x 32 k) into LDS ----
#if ATHENA_TDM
        if (MT == 4) {
            // Tensor Data Mover: one 2D tile (rows=MT*16, cols=32 bf16, stride=Kpad)
            if (wave == 0) {
                unsigned long long ga = gbase + (unsigned long long)ks * 2ull;
                u32x4 g0;
                g0[0] = 1u;                                   // count=1, user mode
                g0[1] = ldsAddr;                              // lds_addr (bytes)
                g0[2] = (unsigned)ga;                         // global_addr[31:0]
                g0[3] = (unsigned)((ga >> 32) & 0x1FFFFFFu)   // global_addr[56:32]
                      | 0x80000000u;                          // type=2 (image)
                // group1 (256b) assembled as 4x u64
                unsigned long long q0 = (1ull << 16)                        // data_size=2B
                                      | ((unsigned long long)(Kpad & 0xFFFF) << 48);
                unsigned long long q1 = (unsigned long long)((unsigned)Kpad >> 16)
                                      | ((unsigned long long)(MT * 16) << 16)  // tensor_dim1
                                      | (32ull << 48);                          // tile_dim0
                unsigned long long q2 = (unsigned long long)(MT * 16)           // tile_dim1
                                      | ((unsigned long long)(unsigned)Kpad << 32); // dim0_stride
                unsigned long long q3 = 0ull;
                i32x8 g1;
                g1[0] = (int)(unsigned)q0; g1[1] = (int)(unsigned)(q0 >> 32);
                g1[2] = (int)(unsigned)q1; g1[3] = (int)(unsigned)(q1 >> 32);
                g1[4] = (int)(unsigned)q2; g1[5] = (int)(unsigned)(q2 >> 32);
                g1[6] = (int)(unsigned)q3; g1[7] = (int)(unsigned)(q3 >> 32);
                i32x4 gz = {0, 0, 0, 0};
#if __clang_major__ >= 23
                i32x8 gz8 = {0, 0, 0, 0, 0, 0, 0, 0};
                __builtin_amdgcn_tensor_load_to_lds(g0, g1, gz, gz, gz8, 0);
#else
                __builtin_amdgcn_tensor_load_to_lds(g0, g1, gz, gz, 0);
#endif
                __builtin_amdgcn_s_wait_tensorcnt(0);
            }
        } else
#endif
        {
#pragma unroll
            for (int m = 0; m < MT; ++m) {
                const unsigned short* gsrc = A + (size_t)(m * 16 + ar) * Kpad + ks + ak;
                unsigned short* ldst = &sA[(m * 16 + ar) * 32 + ak];
#if ATHENA_ASYNC_LDS
                __builtin_amdgcn_global_load_async_to_lds_b32(
                    (GInt*)(gsrc), (LInt*)(ldst), 0, 0);
#else
                *(unsigned*)ldst = *(const unsigned*)gsrc;
#endif
            }
#if ATHENA_ASYNC_LDS
            __builtin_amdgcn_s_wait_asynccnt(0);
#endif
        }
        if (ks + 32 < Kpad)
            __builtin_prefetch(A + (size_t)ar * Kpad + ks + 32 + ak, 0, 1);
        __syncthreads();

        // gather B fragment: k -> (ci, kx, ky, kz) by shifts (pow2 layout)
        Frag16 b;
#pragma unroll
        for (int v = 0; v < 8; ++v) {
            int k = kk[v];
#pragma unroll
            for (int e = 0; e < 2; ++e) {
                int kg = ks + k + e;
                int ci = kg >> 9;
                int kx = (kg >> 6) & 7, ky = (kg >> 3) & 7, kz = kg & 7;
                unsigned short bv = 0;
                if (((unsigned)(idb + kx) < (unsigned)ID) &
                    ((unsigned)(ihb + ky) < (unsigned)IH) &
                    ((unsigned)(iwb + kz) < (unsigned)IW)) {
                    bv = ext[base + ci * IDHW + kx * IHIW + ky * IW + kz];
                }
                b.u[2 * v + e] = bv;
            }
        }

        // MT matrix ops reuse the single B fragment
#pragma unroll
        for (int m = 0; m < MT; ++m) {
            Frag16 a;
#pragma unroll
            for (int v = 0; v < 8; ++v) {
                int k = kk[v];
                a.u[2 * v]     = sA[(m * 16 + row) * 32 + k];
                a.u[2 * v + 1] = sA[(m * 16 + row) * 32 + k + 1];
            }
            c[m] = __builtin_amdgcn_wmma_f32_16x16x32_bf16(false, a.v, false, b.v,
                                                           (short)0, c[m], false, false);
        }
        __syncthreads();
    }

    // D layout: VGPR r -> M = r (+8 for lanes 16..31), N = lane%16
    const int dRow0 = (lane < 16) ? 0 : 8;
#pragma unroll
    for (int m = 0; m < MT; ++m)
#pragma unroll
        for (int r = 0; r < 8; ++r)
            y[(size_t)(m * 16 + dRow0 + r) * Nvox + (size_t)(voxBase + col)] = c[m][r];
}

// ---------- per-channel stats over y rows ----------
__global__ __launch_bounds__(256) void block_stats(const float* __restrict__ y, float* st,
                                                   int so, int vo, int Nvox) {
    __shared__ float s1[256], s2[256];
    int tid = threadIdx.x, c = blockIdx.x;
    float a = 0.f, b = 0.f;
    if (c < so) {
        const float* row = y + (size_t)c * Nvox;
        for (int i = tid; i < Nvox; i += 256) { float v = row[i]; a += v; b += v * v; }
    } else {
        int v = c - so;
        for (int rr = 0; rr < 3; rr++) {
            const float* row = y + (size_t)(so + 3 * v + rr) * Nvox;
            for (int i = tid; i < Nvox; i += 256) { float t = row[i]; a += t * t; }
        }
    }
    s1[tid] = a; s2[tid] = b;
    __syncthreads();
    for (int s = 128; s > 0; s >>= 1) {
        if (tid < s) { s1[tid] += s1[tid + s]; s2[tid] += s2[tid + s]; }
        __syncthreads();
    }
    if (tid == 0) {
        if (c < so) { float m = s1[0] / (float)Nvox; st[c] = m; st[64 + c] = s2[0] / (float)Nvox - m * m; }
        else        { st[128 + (c - so)] = s1[0] / (float)Nvox; }
    }
}

// ---------- norm + ReLU (scalars) / RMS gate (vectors) ----------
__global__ __launch_bounds__(256) void finalize_block(const float* __restrict__ y,
                                                      const float* __restrict__ st,
                                                      const float* __restrict__ gs,
                                                      const float* __restrict__ gv,
                                                      const float* __restrict__ bias,
                                                      unsigned short* __restrict__ actOut,
                                                      float* __restrict__ fOut, int writeFloat,
                                                      int B, int so, int vo, int ODHW, int Nvox) {
    int Cnext = so + 3 * vo;
    int total = B * Cnext * ODHW;
    for (int idx = blockIdx.x * blockDim.x + threadIdx.x; idx < total; idx += gridDim.x * blockDim.x) {
        int vox = idx % ODHW; int t = idx / ODHW;
        int c = t % Cnext; int n = t / Cnext;
        int q = n * ODHW + vox;
        float val;
        if (c < so) {
            float m = st[c], var = st[64 + c];
            val = (y[(size_t)c * Nvox + q] - m) * rsqrtf(var + EPS) * gs[c] + bias[c];
            val = fmaxf(val, 0.f);
        } else {
            int v = (c - so) / 3;
            val = y[(size_t)c * Nvox + q] * rsqrtf(st[128 + v] + EPS) * gv[v];
        }
        if (writeFloat) fOut[idx] = val;
        else            actOut[idx] = f2bf(val);
    }
}

// ---------- pool + batchnorm + linear head ----------
__global__ __launch_bounds__(256) void head(const float* __restrict__ feat,   // [16][64][64]
                                            const float* __restrict__ bn_g,
                                            const float* __restrict__ bn_b,
                                            const float* __restrict__ lin_w,  // [64][10]
                                            const float* __restrict__ lin_b,
                                            float* __restrict__ out) {
    __shared__ float p[16 * 64];
    __shared__ float z[16 * 64];
    int tid = threadIdx.x;
    for (int i = tid; i < 1024; i += 256) {
        int n = i / 64, c = i % 64;
        float s = 0.f;
        for (int v = 0; v < 64; v++) s += feat[(size_t)(n * 64 + c) * 64 + v];
        p[i] = s * (1.f / 64.f);
    }
    __syncthreads();
    for (int c = tid; c < 64; c += 256) {
        float mu = 0.f;
        for (int n = 0; n < 16; n++) mu += p[n * 64 + c];
        mu *= (1.f / 16.f);
        float var = 0.f;
        for (int n = 0; n < 16; n++) { float d = p[n * 64 + c] - mu; var += d * d; }
        var *= (1.f / 16.f);
        float inv = rsqrtf(var + EPS);
        for (int n = 0; n < 16; n++) z[n * 64 + c] = (p[n * 64 + c] - mu) * inv * bn_g[c] + bn_b[c];
    }
    __syncthreads();
    for (int i = tid; i < 160; i += 256) {
        int n = i / 10, j = i % 10;
        float s = lin_b[j];
        for (int c = 0; c < 64; c++) s += z[n * 64 + c] * lin_w[c * 10 + j];
        out[i] = s;
    }
}

// ======================= host orchestration =======================
struct BlkCfg { int Cin, Cout, CoutPad, ID, OD, Kpad, Nvox, si, vi, so, vo; };
static const BlkCfg CFG[4] = {
    { 1, 20, 32, 64, 32,   512, 524288,  1, 0,  8, 4},
    {56, 20, 32, 32, 16, 28672,  65536,  8, 4,  8, 4},
    {56, 28, 32, 16,  8, 28672,   8192,  8, 4, 16, 4},
    {64, 64, 64,  8,  4, 32768,   1024, 16, 4, 64, 0},
};

struct WDesc { int bi, wIdx, mo, dO, mi, dI, nb, row0, col0; unsigned seed; };
static const WDesc WD[17] = {
    {0,  1,  8, 1,  1, 1, 3,  0,  0,    0},
    {0,  2,  4, 3,  1, 1, 3,  8,  0,   10},
    {1,  6,  8, 1,  8, 1, 3,  0,  0, 1000},
    {1,  7,  8, 1,  4, 3, 3,  0,  8, 1001},
    {1,  8,  8, 1,  4, 9, 3,  0, 20, 1002},
    {1,  9,  4, 3,  8, 1, 3,  8,  0, 1010},
    {1, 10,  4, 3,  4, 3, 9,  8,  8, 1011},
    {1, 11,  4, 3,  4, 9, 9,  8, 20, 1012},
    {2, 15, 16, 1,  8, 1, 3,  0,  0, 2000},
    {2, 16, 16, 1,  4, 3, 3,  0,  8, 2001},
    {2, 17, 16, 1,  4, 9, 3,  0, 20, 2002},
    {2, 18,  4, 3,  8, 1, 3, 16,  0, 2010},
    {2, 19,  4, 3,  4, 3, 9, 16,  8, 2011},
    {2, 20,  4, 3,  4, 9, 9, 16, 20, 2012},
    {3, 24, 64, 1, 16, 1, 3,  0,  0, 3000},
    {3, 25, 64, 1,  4, 3, 3,  0, 16, 3001},
    {3, 26, 64, 1,  4, 9, 3,  0, 28, 3002},
};

extern "C" void kernel_launch(void* const* d_in, const int* in_sizes, int n_in,
                              void* d_out, int out_size, void* d_ws, size_t ws_size,
                              hipStream_t stream) {
    (void)in_sizes; (void)n_in; (void)out_size; (void)ws_size;
    char* ws = (char*)d_ws;
    size_t off = 0;
    auto alloc = [&](size_t bytes) -> size_t {
        size_t o = off; off += (bytes + 255) & ~(size_t)255; return o;
    };

    size_t statsOff  = alloc(256);
    size_t bstatsOff = alloc(256 * sizeof(float));
    size_t basisOff[17];
    for (int i = 0; i < 17; i++)
        basisOff[i] = alloc((size_t)WD[i].nb * WD[i].dO * WD[i].dI * 343 * sizeof(float));
    static const size_t kernSize[4] = {6860, 384160, 537824, 1404928};
    size_t kernOff[4];
    for (int b = 0; b < 4; b++) kernOff[b] = alloc(kernSize[b] * sizeof(float));
    size_t aOff[4];
    for (int b = 0; b < 4; b++)
        aOff[b] = alloc((size_t)CFG[b].CoutPad * CFG[b].Kpad * sizeof(unsigned short));
    size_t extOff  = alloc((size_t)29360128 * sizeof(unsigned short)); // max ext (block1)
    size_t actOff  = alloc((size_t)10485760 * sizeof(unsigned short)); // max act (block0 out)
    size_t yOff    = alloc((size_t)16777216 * sizeof(float));          // max y (block0, padded)
    size_t featOff = alloc((size_t)65536 * sizeof(float));             // block3 fp32 out

    float*          stats = (float*)(ws + statsOff);
    float*          bst   = (float*)(ws + bstatsOff);
    unsigned short* extB  = (unsigned short*)(ws + extOff);
    unsigned short* actB  = (unsigned short*)(ws + actOff);
    float*          yB    = (float*)(ws + yOff);
    float*          featB = (float*)(ws + featOff);

    const float* x = (const float*)d_in[0];
    const int Nx = 16 * 64 * 64 * 64;

    // 1) input norm -> bf16 NCDHW (block0 ext, Cin=1)
    zero_f<<<1, 64, 0, stream>>>(stats, 2);
    reduce_sum_sq<<<1024, 256, 0, stream>>>(x, stats, Nx);
    normalize_x<<<4096, 256, 0, stream>>>(x, stats, extB, Nx);

    // 2) bases (deterministic MT19937 replication) + kernel assembly + bf16 A layout
    for (int i = 0; i < 17; i++) {
        const WDesc& w = WD[i];
        int total = w.nb * w.dO * w.dI * 343;
        float scale = 1.0f / sqrtf((float)(w.nb * w.dI * 343));
        gen_basis<<<1, 32, 0, stream>>>(w.seed, total, scale, (float*)(ws + basisOff[i]));
    }
    for (int i = 0; i < 17; i++) {
        const WDesc& w = WD[i];
        int total = w.mo * w.dO * w.mi * w.dI * 343;
        assemble<<<(total + 255) / 256, 256, 0, stream>>>(
            (const float*)d_in[w.wIdx], (const float*)(ws + basisOff[i]),
            (float*)(ws + kernOff[w.bi]),
            w.mo, w.dO, w.mi, w.dI, w.nb, w.row0, w.col0, CFG[w.bi].Cin);
    }
    for (int b = 0; b < 4; b++) {
        int total = CFG[b].CoutPad * CFG[b].Kpad;
        kern_to_bf16<<<(total + 255) / 256, 256, 0, stream>>>(
            (const float*)(ws + kernOff[b]), (unsigned short*)(ws + aOff[b]),
            CFG[b].Cout, CFG[b].Cin, CFG[b].Kpad, CFG[b].CoutPad);
    }

    static const int GSI[4] = {3, 12, 21, 27};
    static const int GVI[4] = {4, 13, 22, -1};
    static const int BII[4] = {5, 14, 23, 28};

    // 3) four conv blocks
    for (int b = 0; b < 4; b++) {
        const BlkCfg& C = CFG[b];
        int ODHW = C.OD * C.OD * C.OD;
        int IDHW = C.ID * C.ID * C.ID;
        if (b > 0) {
            int Cprev = CFG[b - 1].so + 3 * CFG[b - 1].vo;
            int total = 16 * C.Cin * IDHW;
            build_ext<<<(total + 255) / 256, 256, 0, stream>>>(
                actB, extB, 16, Cprev, C.si, C.vi, IDHW);
        }
        dim3 g(C.Nvox / 128);
        if (b < 3)
            conv_wmma<2><<<g, 256, 0, stream>>>(
                extB, (const unsigned short*)(ws + aOff[b]), yB,
                C.Cin, C.ID, C.ID, C.ID, C.OD, C.OD, C.OD, C.Kpad, C.Nvox);
        else
            conv_wmma<4><<<g, 256, 0, stream>>>(
                extB, (const unsigned short*)(ws + aOff[b]), yB,
                C.Cin, C.ID, C.ID, C.ID, C.OD, C.OD, C.OD, C.Kpad, C.Nvox);
        block_stats<<<C.so + C.vo, 256, 0, stream>>>(yB, bst, C.so, C.vo, C.Nvox);
        int Cnext = C.so + 3 * C.vo;
        int total = 16 * Cnext * ODHW;
        finalize_block<<<(total + 255) / 256, 256, 0, stream>>>(
            yB, bst,
            (const float*)d_in[GSI[b]],
            (b < 3) ? (const float*)d_in[GVI[b]] : (const float*)d_in[GSI[b]],
            (const float*)d_in[BII[b]],
            actB, featB, (b == 3) ? 1 : 0,
            16, C.so, C.vo, ODHW, C.Nvox);
    }

    // 4) pool + batchnorm + linear
    head<<<1, 256, 0, stream>>>(featB,
                                (const float*)d_in[29], (const float*)d_in[30],
                                (const float*)d_in[31], (const float*)d_in[32],
                                (float*)d_out);
}
